// OneGNN_37177236914930
// MI455X (gfx1250) — matrix operand order repaired and checked by reading, not compiled, not run on hardware
//
#include <hip/hip_runtime.h>

// ---------------------------------------------------------------------------
// GIN 3-layer forward for MI455X (gfx1250), wave32 + WMMA bf16.
//   per layer: scatter-add over edges (explicit global_atomic_add_f32,
//   L2-resident x), then one fused kernel: h=(1+eps)x+agg -> GEMM1+BN+ReLU
//   -> GEMM2+BN+ReLU -> outer BN (+ReLU), staged through LDS, GEMMs via
//   v_wmma_f32_16x16x32_bf16 (bf16 in, f32 accumulate).
// ---------------------------------------------------------------------------

typedef __attribute__((ext_vector_type(16))) __bf16 v16bf;
typedef __attribute__((ext_vector_type(8)))  float  v8f;

#define NNODES   100000
#define NEDGES   1600000
#define CCH      64
#define INV_STD  0.9999950000374997f   // 1/sqrt(1 + 1e-5)

__device__ __forceinline__ unsigned short f2bf_bits(float f) {
    __bf16 b = (__bf16)f;
    return __builtin_bit_cast(unsigned short, b);
}

// Guaranteed-native f32 atomic add: saddr form, no return (STOREcnt-tracked),
// device scope so the next kernel in the stream observes the sums.
__device__ __forceinline__ void atomic_add_f32_dev(float* base, int byteoff, float v) {
    asm volatile("global_atomic_add_f32 %0, %1, %2 scope:SCOPE_DEV"
                 :
                 : "v"(byteoff), "v"(v), "s"(base)
                 : "memory");
}

// --------------------------- weight pre-pack -------------------------------
// Packs W[64][64] (row-major [k][n]) into the CDNA5 WMMA bf16 B-operand lane
// layout: for output-col tile nb (4) and K-block kb (2, K=32 each), lane L
// (n = L%16, hi = L/16) holds VGPR v pairs K = kb*32 + (v<4 ? 2v : 16+2(v-4)) + 8*hi.
// Layout: pk[l][mat][nb][kb][lane][16 bf16]  (contiguous 32B per lane).
__global__ void pack_w_kernel(const float* __restrict__ W1,
                              const float* __restrict__ W2,
                              unsigned short* __restrict__ pk) {
    int t = blockIdx.x * blockDim.x + threadIdx.x;
    if (t >= 3 * 2 * 4 * 2 * 32) return;
    int lane = t & 31;
    int kb   = (t >> 5) & 1;
    int nb   = (t >> 6) & 3;
    int mat  = (t >> 8) & 1;
    int l    = t >> 9;
    const float* W = (mat ? W2 : W1) + l * 64 * 64;
    int n  = nb * 16 + (lane & 15);
    int hi = lane >> 4;
    unsigned short* o = pk + (size_t)t * 16;
#pragma unroll
    for (int v = 0; v < 8; ++v) {
        int k = kb * 32 + (v < 4 ? 2 * v : 8 + 2 * v) + 8 * hi;
        o[2 * v]     = f2bf_bits(W[k * 64 + n]);
        o[2 * v + 1] = f2bf_bits(W[(k + 1) * 64 + n]);
    }
}

// ------------------------------ zero agg -----------------------------------
__global__ void zero_kernel(float4* __restrict__ p, int n4) {
    int i = blockIdx.x * blockDim.x + threadIdx.x;
    int stride = gridDim.x * blockDim.x;
    for (; i < n4; i += stride) {
        float4 z; z.x = 0.f; z.y = 0.f; z.z = 0.f; z.w = 0.f;
        p[i] = z;
    }
}

// --------------------------- edge scatter-add ------------------------------
// One wave per edge: 32 lanes * float2 = the full 256B feature row (coalesced
// gather, L2 hit since x is 25.6MB << 192MB L2), then 2 native f32 atomics
// per lane into agg[dst].
__global__ void __launch_bounds__(256)
scatter_kernel(const float* __restrict__ x, const int* __restrict__ src,
               const int* __restrict__ dst, float* __restrict__ agg) {
    int e = blockIdx.x * 8 + (threadIdx.x >> 5);
    if (e >= NEDGES) return;
    int lane = threadIdx.x & 31;
    int s = src[e];
    int d = dst[e];
    const float2 v = *(const float2*)(x + (size_t)s * CCH + lane * 2);
    int boff = (d * CCH + lane * 2) * 4;   // byte offset, < 25.6MB -> fits i32
    atomic_add_f32_dev(agg, boff,     v.x);
    atomic_add_f32_dev(agg, boff + 4, v.y);
}

// ------------------------- fused GIN MLP layer -----------------------------
// Block = 256 threads = 8 waves; each wave owns a 16-node tile through LDS
// (row stride 68 floats keeps float4 alignment and the 16-lane column
// accesses at worst 2-way bank-conflicted).
#define LDSTRIDE 68

// Build a bf16 A-operand (16x32 K-block) from an fp32 LDS tile, following the
// ISA A layout: lane (m = lane%16, hi), VGPR v holds K pair
//   k = kbase + (v<4 ? 2v : 16+2(v-4)) + 8*hi.
__device__ __forceinline__ v16bf load_a_tile(const float (*t)[LDSTRIDE],
                                             int m, int hi, int kbase) {
    v16bf a;
#pragma unroll
    for (int v = 0; v < 8; ++v) {
        int k = kbase + (v < 4 ? 2 * v : 8 + 2 * v) + 8 * hi;
        a[2 * v]     = (__bf16)t[m][k];
        a[2 * v + 1] = (__bf16)t[m][k + 1];
    }
    return a;
}

__global__ void __launch_bounds__(256)
gin_mlp_kernel(const float* __restrict__ xin, const float* __restrict__ agg,
               const float* __restrict__ epsp, int layer,
               const unsigned short* __restrict__ pkW1,
               const unsigned short* __restrict__ pkW2,
               const float* __restrict__ b1, const float* __restrict__ g1,
               const float* __restrict__ be1,
               const float* __restrict__ b2, const float* __restrict__ g2,
               const float* __restrict__ be2,
               const float* __restrict__ bng, const float* __restrict__ bnb,
               float* __restrict__ out, int final_relu) {
    __shared__ float tile[8][16][LDSTRIDE];
    const int w    = threadIdx.x >> 5;
    const int lane = threadIdx.x & 31;
    const int m    = lane & 15;
    const int hi   = lane >> 4;
    const int rowbase = blockIdx.x * 128 + w * 16;
    const float ep = 1.0f + epsp[layer];

    // Stage 1: h = (1+eps)*x + agg  ->  LDS tile (coalesced float4)
    {
        const int col4 = m * 4;
#pragma unroll
        for (int it = 0; it < 8; ++it) {
            int r = it * 2 + hi;
            int row = rowbase + r;
            float4 hv; hv.x = 0.f; hv.y = 0.f; hv.z = 0.f; hv.w = 0.f;
            if (row < NNODES) {
                const float4 xv = *(const float4*)(xin + (size_t)row * CCH + col4);
                const float4 av = *(const float4*)(agg + (size_t)row * CCH + col4);
                hv.x = ep * xv.x + av.x;
                hv.y = ep * xv.y + av.y;
                hv.z = ep * xv.z + av.z;
                hv.w = ep * xv.w + av.w;
            }
            *(float4*)&tile[w][r][col4] = hv;
        }
    }
    __syncthreads();

    // Stage 2: GEMM1 (h @ W1) with BN1+ReLU epilogue, back into LDS.
    {
        const v16bf a0 = load_a_tile(tile[w], m, hi, 0);
        const v16bf a1 = load_a_tile(tile[w], m, hi, 32);
        __syncthreads();
#pragma unroll
        for (int nb = 0; nb < 4; ++nb) {
            const v16bf bb0 = *(const v16bf*)(pkW1 + (size_t)((nb * 2 + 0) * 32 + lane) * 16);
            const v16bf bb1 = *(const v16bf*)(pkW1 + (size_t)((nb * 2 + 1) * 32 + lane) * 16);
            v8f acc = {};
            acc = __builtin_amdgcn_wmma_f32_16x16x32_bf16(false, a0, false, bb0,
                                                          (short)0, acc, false, false);
            acc = __builtin_amdgcn_wmma_f32_16x16x32_bf16(false, a1, false, bb1,
                                                          (short)0, acc, false, false);
            const int col = nb * 16 + m;
            const float sc = g1[col] * INV_STD;
            const float sh = sc * b1[col] + be1[col];
#pragma unroll
            for (int r = 0; r < 8; ++r) {
                float y = acc[r] * sc + sh;
                tile[w][r + 8 * hi][col] = fmaxf(y, 0.0f);
            }
        }
    }
    __syncthreads();

    // Stage 3: GEMM2 (y1 @ W2) with BN2+ReLU and outer BN (+ReLU) epilogue.
    {
        const v16bf c0 = load_a_tile(tile[w], m, hi, 0);
        const v16bf c1 = load_a_tile(tile[w], m, hi, 32);
        __syncthreads();
#pragma unroll
        for (int nb = 0; nb < 4; ++nb) {
            const v16bf bb0 = *(const v16bf*)(pkW2 + (size_t)((nb * 2 + 0) * 32 + lane) * 16);
            const v16bf bb1 = *(const v16bf*)(pkW2 + (size_t)((nb * 2 + 1) * 32 + lane) * 16);
            v8f acc = {};
            acc = __builtin_amdgcn_wmma_f32_16x16x32_bf16(false, c0, false, bb0,
                                                          (short)0, acc, false, false);
            acc = __builtin_amdgcn_wmma_f32_16x16x32_bf16(false, c1, false, bb1,
                                                          (short)0, acc, false, false);
            const int col = nb * 16 + m;
            const float sc2 = g2[col] * INV_STD;
            const float sh2 = sc2 * b2[col] + be2[col];
            const float so  = bng[col] * INV_STD;
            const float bo  = bnb[col];
#pragma unroll
            for (int r = 0; r < 8; ++r) {
                float y = fmaxf(acc[r] * sc2 + sh2, 0.0f);
                float z = so * y + bo;
                if (final_relu) z = fmaxf(z, 0.0f);
                tile[w][r + 8 * hi][col] = z;
            }
        }
    }
    __syncthreads();

    // Stage 4: coalesced float4 store of the finished tile.
    {
        const int col4 = m * 4;
#pragma unroll
        for (int it = 0; it < 8; ++it) {
            int r = it * 2 + hi;
            int row = rowbase + r;
            if (row < NNODES)
                *(float4*)(out + (size_t)row * CCH + col4) = *(const float4*)&tile[w][r][col4];
        }
    }
}

// ------------------------------- launcher ----------------------------------
extern "C" void kernel_launch(void* const* d_in, const int* in_sizes, int n_in,
                              void* d_out, int out_size, void* d_ws, size_t ws_size,
                              hipStream_t stream) {
    const float* x   = (const float*)d_in[0];
    const int*   ei  = (const int*)d_in[1];
    const float* eps = (const float*)d_in[2];
    const float* W1  = (const float*)d_in[3];
    const float* b1  = (const float*)d_in[4];
    const float* g1  = (const float*)d_in[5];
    const float* be1 = (const float*)d_in[6];
    const float* W2  = (const float*)d_in[7];
    const float* b2  = (const float*)d_in[8];
    const float* g2  = (const float*)d_in[9];
    const float* be2 = (const float*)d_in[10];
    const float* bng = (const float*)d_in[11];
    const float* bnb = (const float*)d_in[12];
    float* out = (float*)d_out;

    // workspace: agg (25.6MB) | ping buffer (25.6MB) | packed bf16 weights (48KB)
    char* ws = (char*)d_ws;
    float* agg            = (float*)ws;
    float* bufA           = (float*)(ws + (size_t)NNODES * CCH * 4);
    unsigned short* pk    = (unsigned short*)(ws + 2 * (size_t)NNODES * CCH * 4);

    pack_w_kernel<<<6, 256, 0, stream>>>(W1, W2, pk);

    const int n4 = NNODES * CCH / 4;
    const int mlp_blocks = (NNODES + 127) / 128;
    const int scat_blocks = (NEDGES + 7) / 8;

    const float* xi = x;
    float* outs[3] = { out, bufA, out };   // ping-pong, final layer -> d_out
    for (int l = 0; l < 3; ++l) {
        zero_kernel<<<2048, 256, 0, stream>>>((float4*)agg, n4);
        scatter_kernel<<<scat_blocks, 256, 0, stream>>>(xi, ei, ei + NEDGES, agg);
        gin_mlp_kernel<<<mlp_blocks, 256, 0, stream>>>(
            xi, agg, eps, l,
            pk + (size_t)(l * 2 + 0) * 4096, pk + (size_t)(l * 2 + 1) * 4096,
            b1 + l * 64, g1 + l * 64, be1 + l * 64,
            b2 + l * 64, g2 + l * 64, be2 + l * 64,
            bng + l * 64, bnb + l * 64,
            outs[l], (l < 2) ? 1 : 0);
        xi = outs[l];
    }
}
